// GCNAutoencoder_36945308680357
// MI455X (gfx1250) — matrix operand order repaired and checked by reading, not compiled, not run on hardware
//
#include <hip/hip_runtime.h>

typedef __attribute__((ext_vector_type(16))) _Float16 v16h;
typedef __attribute__((ext_vector_type(8)))  _Float16 v8h;
typedef __attribute__((ext_vector_type(4)))  _Float16 v4h;
typedef __attribute__((ext_vector_type(2)))  _Float16 v2h;
typedef __attribute__((ext_vector_type(8)))  float    v8f;

#define N_NODES 50000
#define NFEAT   512
#define HID     128
#define NEDGE   1600000

#define MT 5            // 16-row output tiles per block
#define BM (MT * 16)    // 80 rows per block; 50000 = 80 * 625 exactly
#define LPAD 40         // LDS row pitch in halves: 80B stride, conflict-free

// ---------------------------------------------------------------------------
// WMMA GEMM: C[M,N] = A[M,K] @ B[K,N] (+bias, ReLU if EPI), f16 mul, f32 acc.
// Block = 256 threads (8 waves) computes an 80(M) x 128(N) strip.
// Wave w owns columns [n0+16w, n0+16w+16); 5 WMMAs (M tiles) per k-step.
// B tile staged TRANSPOSED in LDS so B fragments are contiguous b128 loads.
// ---------------------------------------------------------------------------
template <int K, int N, bool EPI>
__global__ void __launch_bounds__(256)
gemm_wmma(const float* __restrict__ A, const float* __restrict__ B,
          float* __restrict__ C, const float* __restrict__ bias) {
  __shared__ _Float16 As[BM][LPAD];     // 80x32 used  (6.4 KB)
  __shared__ _Float16 BsT[128][LPAD];   // [n][k], 32 used (10.2 KB)

  const int tid  = threadIdx.x;
  const int lane = tid & 31;
  const int wave = tid >> 5;
  const int m0 = blockIdx.x * BM;
  const int n0 = blockIdx.y * 128;

  // Fragment addressing (this code's contiguous k order, same for A and B):
  // A 16x32 f16: lane holds row M=lane&15, K sequential from 16*(lane>=16)
  // B 32x16 f16: lane holds col N=lane&15, K sequential from 16*(lane>=16)
  const int mrow_a = lane & 15;
  const int kb_a   = (lane >> 4) * 16;
  const int ncol_b = wave * 16 + (lane & 15);
  const int kb_b   = (lane >> 4) * 16;

  v8f c[MT] = {};

  for (int k0 = 0; k0 < K; k0 += 32) {
    // ---- stage A tile (80x32 f32 -> f16): 1280 float2, 5 per thread
#pragma unroll
    for (int j = 0; j < MT; ++j) {
      int idx = tid + j * 256;
      int r  = idx >> 4;         // 16 float2 per row
      int c2 = idx & 15;
      const float2 t = *(const float2*)&A[(size_t)(m0 + r) * K + k0 + c2 * 2];
      v2h p; p[0] = (_Float16)t.x; p[1] = (_Float16)t.y;
      *(v2h*)&As[r][c2 * 2] = p;
    }
    // ---- stage B tile (32x128) transposed: thread packs 4 K x 1 N -> b64
#pragma unroll
    for (int j = 0; j < 4; ++j) {
      int idx = tid + j * 256;       // 0..1023
      int nc  = idx & 127;           // coalesced across threads
      int kg  = (idx >> 7) * 4;      // 0,4,...,28
      v4h p;
#pragma unroll
      for (int i = 0; i < 4; ++i)
        p[i] = (_Float16)B[(size_t)(k0 + kg + i) * N + n0 + nc];
      *(v4h*)&BsT[nc][kg] = p;
    }
    __syncthreads();

    // B fragment: 16 contiguous halves -> 2x ds_load_b128
    const v8h* bp = (const v8h*)&BsT[ncol_b][kb_b];
    const v8h b_lo = bp[0], b_hi = bp[1];
    const v16h b = __builtin_shufflevector(b_lo, b_hi,
        0,1,2,3,4,5,6,7,8,9,10,11,12,13,14,15);

#pragma unroll
    for (int mt = 0; mt < MT; ++mt) {
      // A fragment: halves [kb, kb+16) -> 2x ds_load_b128
      const v8h* ap = (const v8h*)&As[mt * 16 + mrow_a][kb_a];
      const v8h a_lo = ap[0], a_hi = ap[1];
      const v16h a = __builtin_shufflevector(a_lo, a_hi,
          0,1,2,3,4,5,6,7,8,9,10,11,12,13,14,15);
      c[mt] = __builtin_amdgcn_wmma_f32_16x16x32_f16(false, a, false, b,
                                                     (short)0, c[mt], false, false);
    }
    __syncthreads();
  }

  // D layout: VGPR r -> M = 8*(lane>=16) + r, N = 16*wave + (lane&15)
  const int mo = m0 + (lane >> 4) * 8;
  const int no = n0 + wave * 16 + (lane & 15);
  const float bv = EPI ? bias[no] : 0.0f;   // per-column bias, fixed per lane
#pragma unroll
  for (int mt = 0; mt < MT; ++mt)
#pragma unroll
    for (int r = 0; r < 8; ++r) {
      float t = c[mt][r];
      if (EPI) { t += bv; t = t > 0.0f ? t : 0.0f; }
      C[(size_t)(mo + mt * 16 + r) * N + no] = t;
    }
}

// ---------------------------------------------------------------------------
// Edge-parallel SpMM: out[row[e], :] += val[e] * dense[col[e], :]
// One wave per edge (grid-stride): b128 gathers from the L2-resident table,
// non-returning f32 scatter-atomics (RMW performed in L2).
// ---------------------------------------------------------------------------
__global__ void __launch_bounds__(256)
spmm_atomic(const int* __restrict__ row, const int* __restrict__ col,
            const float* __restrict__ val, const float* __restrict__ dense,
            float* __restrict__ out, int E, int D) {
  const int lane   = threadIdx.x & 31;
  const int wave   = (int)((blockIdx.x * blockDim.x + threadIdx.x) >> 5);
  const int nwaves = (int)((gridDim.x * blockDim.x) >> 5);
  const int d4 = D >> 2;
  for (int e = wave; e < E; e += nwaves) {
    const int   r  = row[e];
    const int   cc = col[e];
    const float v  = val[e];
    const float4* __restrict__ src = (const float4*)(dense + (size_t)cc * D);
    float* dst = out + (size_t)r * D;
    for (int i = lane; i < d4; i += 32) {
      float4 x = src[i];
      unsafeAtomicAdd(dst + 4 * i + 0, v * x.x);
      unsafeAtomicAdd(dst + 4 * i + 1, v * x.y);
      unsafeAtomicAdd(dst + 4 * i + 2, v * x.z);
      unsafeAtomicAdd(dst + 4 * i + 3, v * x.w);
    }
  }
}

// ---------------------------------------------------------------------------
__global__ void __launch_bounds__(256)
bias_relu(const float* __restrict__ acc, const float* __restrict__ bias,
          float* __restrict__ out, size_t total, int Dmask) {
  size_t i = (size_t)blockIdx.x * blockDim.x + threadIdx.x;
  size_t stride = (size_t)gridDim.x * blockDim.x;
  for (; i < total; i += stride) {
    float t = acc[i] + bias[i & Dmask];
    out[i] = t > 0.0f ? t : 0.0f;
  }
}

__global__ void __launch_bounds__(256)
zero_f32(float4* __restrict__ p, size_t n4) {
  size_t i = (size_t)blockIdx.x * blockDim.x + threadIdx.x;
  size_t stride = (size_t)gridDim.x * blockDim.x;
  float4 z = {0.f, 0.f, 0.f, 0.f};
  for (; i < n4; i += stride) p[i] = z;
}

// ---------------------------------------------------------------------------
extern "C" void kernel_launch(void* const* d_in, const int* in_sizes, int n_in,
                              void* d_out, int out_size, void* d_ws, size_t ws_size,
                              hipStream_t stream) {
  const float* x       = (const float*)d_in[0];   // [N, 512]
  const int*   adj_row = (const int*)  d_in[1];   // [E]
  const int*   adj_col = (const int*)  d_in[2];   // [E]
  const float* adj_val = (const float*)d_in[3];   // [E]
  const int*   inv_row = (const int*)  d_in[4];   // [E]
  const int*   inv_col = (const int*)  d_in[5];   // [E]
  const float* inv_val = (const float*)d_in[6];   // [E]
  const float* W1      = (const float*)d_in[7];   // [512, 128]
  const float* b1      = (const float*)d_in[8];   // [128]
  const float* W2      = (const float*)d_in[9];   // [128, 512]
  const float* b2      = (const float*)d_in[10];  // [512]

  float* out = (float*)d_out;                          // [N, 512]
  float* emb = out + (size_t)N_NODES * NFEAT;          // [N, 128] (output 2)

  // Workspace (peak 51.2 MB):
  //   xW1 [N,128] @ 0   (later reused as g = adj_inv @ emb)
  //   h_pre [N,128] @ +25.6MB
  float* ws    = (float*)d_ws;
  float* xW1   = ws;
  float* h_pre = ws + (size_t)N_NODES * HID;
  float* g     = ws;      // reuses xW1 (dead after spmm #1 completes)

  // 1) xW1 = x @ W1     (M=50000, K=512, N=128)
  gemm_wmma<NFEAT, HID, false>
      <<<dim3(N_NODES / BM, HID / 128), 256, 0, stream>>>(x, W1, xW1, nullptr);

  // 2) h_pre = adj @ xW1          (SpMM over D=128)
  zero_f32<<<2048, 256, 0, stream>>>((float4*)h_pre, (size_t)N_NODES * HID / 4);
  spmm_atomic<<<4096, 256, 0, stream>>>(adj_row, adj_col, adj_val, xW1, h_pre,
                                        NEDGE, HID);

  // 3) emb = relu(h_pre + b1)     (output #2, and the decoder's input)
  bias_relu<<<4096, 256, 0, stream>>>(h_pre, b1, emb, (size_t)N_NODES * HID, HID - 1);

  // 4) g = adj_inv @ emb          (SpMM over D=128 — associativity:
  //    adj_inv @ (emb @ W2) == (adj_inv @ emb) @ W2; 4x less L2 traffic)
  zero_f32<<<2048, 256, 0, stream>>>((float4*)g, (size_t)N_NODES * HID / 4);
  spmm_atomic<<<4096, 256, 0, stream>>>(inv_row, inv_col, inv_val, emb, g,
                                        NEDGE, HID);

  // 5) out = relu(g @ W2 + b2)    (M=50000, K=128, N=512, fused epilogue)
  gemm_wmma<HID, NFEAT, true>
      <<<dim3(N_NODES / BM, NFEAT / 128), 256, 0, stream>>>(g, W2, out, b2);
}
